// MyModel_87522843560382
// MI455X (gfx1250) — matrix operand order repaired and checked by reading, not compiled, not run on hardware
//
#include <hip/hip_runtime.h>
#include <cstdint>
#include <cstddef>

// ---------------------------------------------------------------------------
// Types for WMMA fragments (gfx1250, wave32)
// ---------------------------------------------------------------------------
typedef __attribute__((ext_vector_type(16))) _Float16 v16h;
typedef __attribute__((ext_vector_type(8)))  _Float16 v8h;
typedef __attribute__((ext_vector_type(8)))  float    v8f;

// Problem constants
#define BATCH 256
#define TSEQ  512
#define U1N 258
#define U2N 124
#define U3N 32

// Layer tiling:
//  L1: A = [x (K 0..31, only k=0 live) | h1_{t-1} (K 32..319)]  K=320 (10 kt), N=272 (17 nt)
//  L2: A = [h1_t (K 0..287)            | h2_{t-1} (K 288..415)] K=416 (13 kt), N=128 ( 8 nt)
//  L3: A = [h2_t (K 0..127)            | h3_{t-1} (K 128..159)] K=160 ( 5 kt), N= 32 ( 2 nt)
#define L1_NT 17
#define L1_KT 10
#define L2_NT 8
#define L2_KT 13
#define L3_NT 2
#define L3_KT 5

#define H1_STRIDE 288   // f16 elements per row in h1 sequence
#define H2_STRIDE 128
#define H3_STRIDE 32
#define C1_STRIDE 272
#define C2_STRIDE 128
#define C3_STRIDE 32

// packed-weight element counts (halves)
#define PACK1_ELEMS (4 * L1_NT * L1_KT * 512)   // 348160
#define PACK2_ELEMS (4 * L2_NT * L2_KT * 512)   // 212992
#define PACK3_ELEMS (4 * L3_NT * L3_KT * 512)   //  20480

// ---------------------------------------------------------------------------
// Fast activations (hardware rcp + exp; values are gate pre-activations,
// bounded by f32 accumulation of f16 products — clamp tanh for safety)
// ---------------------------------------------------------------------------
__device__ __forceinline__ float frcp(float x) { return __builtin_amdgcn_rcpf(x); }
__device__ __forceinline__ float fsigmoid(float x) {
    return frcp(1.0f + __expf(-x));
}
__device__ __forceinline__ float ftanh(float x) {
    x = fminf(15.0f, fmaxf(-15.0f, x));
    float e = __expf(-2.0f * x);
    return (1.0f - e) * frcp(1.0f + e);
}

// ---------------------------------------------------------------------------
// Weight packing: store B-matrix (32K x 16N f16) tiles in this code's B-fragment (contiguous k, same as A)
// lane order so the step kernel loads each lane's 16 halves contiguously.
// lane L (0..31): N = L&15, K = (L>>4)*16 + j, j = 0..15.
// Block layout: [gate][ntile][ktile][lane][16 halves], 512 halves per block.
// ---------------------------------------------------------------------------
__global__ void pack_l1(const float* __restrict__ W1, const float* __restrict__ Ur,
                        _Float16* __restrict__ pack) {
    for (int e = blockIdx.x * blockDim.x + threadIdx.x; e < PACK1_ELEMS;
         e += gridDim.x * blockDim.x) {
        int j = e & 15, lane = (e >> 4) & 31, blk = e >> 9;
        int kt = blk % L1_KT; int r = blk / L1_KT;
        int nt = r % L1_NT;   int g = r / L1_NT;
        int k = kt * 32 + (lane >> 4) * 16 + j;
        int u = nt * 16 + (lane & 15);
        float v = 0.0f;
        if (u < U1N) {
            int col = g * U1N + u;
            if (k == 0)                         v = W1[col];
            else if (k >= 32 && k < 32 + U1N)   v = Ur[(size_t)(k - 32) * (4 * U1N) + col];
        }
        pack[e] = (_Float16)v;
    }
}

__global__ void pack_l2(const float* __restrict__ Wk, const float* __restrict__ Ur,
                        _Float16* __restrict__ pack) {
    for (int e = blockIdx.x * blockDim.x + threadIdx.x; e < PACK2_ELEMS;
         e += gridDim.x * blockDim.x) {
        int j = e & 15, lane = (e >> 4) & 31, blk = e >> 9;
        int kt = blk % L2_KT; int r = blk / L2_KT;
        int nt = r % L2_NT;   int g = r / L2_NT;
        int k = kt * 32 + (lane >> 4) * 16 + j;
        int u = nt * 16 + (lane & 15);
        float v = 0.0f;
        if (u < U2N) {
            int col = g * U2N + u;
            if (k < U1N)                          v = Wk[(size_t)k * (4 * U2N) + col];
            else if (k >= 288 && k < 288 + U2N)   v = Ur[(size_t)(k - 288) * (4 * U2N) + col];
        }
        pack[e] = (_Float16)v;
    }
}

__global__ void pack_l3(const float* __restrict__ Wk, const float* __restrict__ Ur,
                        _Float16* __restrict__ pack) {
    for (int e = blockIdx.x * blockDim.x + threadIdx.x; e < PACK3_ELEMS;
         e += gridDim.x * blockDim.x) {
        int j = e & 15, lane = (e >> 4) & 31, blk = e >> 9;
        int kt = blk % L3_KT; int r = blk / L3_KT;
        int nt = r % L3_NT;   int g = r / L3_NT;
        int k = kt * 32 + (lane >> 4) * 16 + j;
        int u = nt * 16 + (lane & 15);
        float v = 0.0f;
        {
            int col = g * U3N + u;
            if (k < U2N)                          v = Wk[(size_t)k * (4 * U3N) + col];
            else if (k >= 128 && k < 128 + U3N)   v = Ur[(size_t)(k - 128) * (4 * U3N) + col];
        }
        pack[e] = (_Float16)v;
    }
}

// Zero a region (16B vectors; byte count is a multiple of 256)
__global__ void zero_ws(uint32_t* __restrict__ p, size_t n_dwords) {
    typedef __attribute__((ext_vector_type(4))) uint32_t v4u;
    size_t n4 = n_dwords >> 2;
    for (size_t i = blockIdx.x * blockDim.x + threadIdx.x; i < n4;
         i += (size_t)gridDim.x * blockDim.x)
        ((v4u*)p)[i] = (v4u){0u, 0u, 0u, 0u};
}

// ---------------------------------------------------------------------------
// One LSTM cell step for a 16(batch) x 16(unit) tile, 4 gate accumulators.
// Branch-free, fully unrolled k-loop: srcA1 always valid (zero pad buffer at
// t == 0), so the scheduler can software-pipeline loads under WMMAs.
//   xptr != nullptr  => layer-1 x special-case for k-tile 0 (only k==0 live)
// ---------------------------------------------------------------------------
__device__ __forceinline__ void lstm_tile(
    int mt, int nt, int U, int NT, int KT, int ktHB,
    const _Float16* __restrict__ srcA0, int strideA0,
    const _Float16* __restrict__ srcA1, int strideA1,
    const _Float16* __restrict__ pack,
    const float* __restrict__ bias,
    float* __restrict__ cbuf, int strideC,
    _Float16* __restrict__ hout, int strideH,
    const float* __restrict__ xptr, int t)
{
    const int lane = threadIdx.x & 31;
    const int lL = lane & 15;        // N column / batch row selector
    const int hL = lane >> 4;        // K-half selector
    const int m0 = mt * 16, n0 = nt * 16;

    v8f acc[4];
#pragma unroll
    for (int g = 0; g < 4; ++g) acc[g] = (v8f)0.0f;

    // Per-lane row bases (A fragment: lane L<16 holds K {0..15}, L>=16
    // holds K {16..31} of its row).
    const _Float16* aRow0 = srcA0
        ? srcA0 + (size_t)(m0 + lL) * strideA0 + hL * 16 : nullptr;
    const _Float16* aRow1 = srcA1 + (size_t)(m0 + lL) * strideA1 + hL * 16;
    const _Float16* bBase = pack + (((size_t)nt * KT) << 9) + lane * 16;
    const size_t gStride = ((size_t)NT * KT) << 9;   // halves per gate plane

#pragma unroll
    for (int kt = 0; kt < KT; ++kt) {
        v16h a;
        if (kt < ktHB && xptr) {
            // Layer-1 x region: only global k==0 carries data (D == 1).
#pragma unroll
            for (int j = 0; j < 16; ++j) a[j] = (_Float16)0.0f;
            if (hL == 0) a[0] = (_Float16)xptr[(size_t)(m0 + lL) * TSEQ + t];
        } else {
            const _Float16* ap = (kt < ktHB)
                ? aRow0 + kt * 32
                : aRow1 + (kt - ktHB) * 32;
            v8h lo = *(const v8h*)ap;        // K halves +0..7   (lane group)
            v8h hi = *(const v8h*)(ap + 8);  // K halves +8..15
#pragma unroll
            for (int j = 0; j < 8; ++j) { a[j] = lo[j]; a[8 + j] = hi[j]; }
        }
        if (kt + 1 < KT)   // warm next weight block (L2-resident after step 0)
            __builtin_prefetch(bBase + ((size_t)(kt + 1) << 9), 0, 1);
#pragma unroll
        for (int g = 0; g < 4; ++g) {
            v16h b = *(const v16h*)(bBase + (size_t)g * gStride + ((size_t)kt << 9));
            acc[g] = __builtin_amdgcn_wmma_f32_16x16x32_f16(
                false, a, false, b, (short)0, acc[g], false, false);
        }
    }

    // Epilogue: lane owns unit u = n0+lL, batch rows m0 + hL*8 + r (r=0..7).
    const int u = n0 + lL;
    float bI = 0.f, bF = 0.f, bG = 0.f, bO = 0.f;
    if (u < U) {
        bI = bias[0 * U + u]; bF = bias[1 * U + u];
        bG = bias[2 * U + u]; bO = bias[3 * U + u];
    }
#pragma unroll
    for (int r = 0; r < 8; ++r) {
        int m = m0 + hL * 8 + r;
        float ig = fsigmoid(acc[0][r] + bI);
        float fg = fsigmoid(acc[1][r] + bF);
        float gg = ftanh   (acc[2][r] + bG);
        float og = fsigmoid(acc[3][r] + bO);
        size_t ci = (size_t)m * strideC + u;
        float cn = fg * cbuf[ci] + ig * gg;
        cbuf[ci] = cn;
        hout[(size_t)m * strideH + u] = (_Float16)(og * ftanh(cn));
    }
}

// ---------------------------------------------------------------------------
// Fused pipelined super-step: layer1@t=s, layer2@t=s-1, layer3@t=s-2.
// 432 waves: [0,272) L1, [272,400) L2, [400,432) L3. Branches are wave-uniform
// so EXEC stays all-ones inside the WMMA path.
// ---------------------------------------------------------------------------
__global__ void __launch_bounds__(256)
lstm_fused_step(int s,
                const float* __restrict__ x,
                const float* __restrict__ b1, const float* __restrict__ b2,
                const float* __restrict__ b3,
                const _Float16* __restrict__ pack1,
                const _Float16* __restrict__ pack2,
                const _Float16* __restrict__ pack3,
                _Float16* __restrict__ h1seq, _Float16* __restrict__ h2seq,
                _Float16* __restrict__ h3buf,
                const _Float16* __restrict__ zeropad,
                float* __restrict__ c1, float* __restrict__ c2,
                float* __restrict__ c3)
{
    int w = (blockIdx.x * blockDim.x + threadIdx.x) >> 5;
    if (w < 272) {                                   // ---- layer 1 ----
        int t = s;
        if (t >= TSEQ) return;
        int nt = w / 16, mt = w % 16;
        const _Float16* hprev = (t > 0)
            ? h1seq + (size_t)(t - 1) * BATCH * H1_STRIDE : zeropad;
        lstm_tile(mt, nt, U1N, L1_NT, L1_KT, /*ktHB=*/1,
                  nullptr, 0, hprev, H1_STRIDE,
                  pack1, b1, c1, C1_STRIDE,
                  h1seq + (size_t)t * BATCH * H1_STRIDE, H1_STRIDE,
                  x, t);
    } else if (w < 400) {                            // ---- layer 2 ----
        int t = s - 1;
        if (t < 0 || t >= TSEQ) return;
        int w2 = w - 272; int nt = w2 / 16, mt = w2 % 16;
        const _Float16* hprev = (t > 0)
            ? h2seq + (size_t)(t - 1) * BATCH * H2_STRIDE : zeropad;
        lstm_tile(mt, nt, U2N, L2_NT, L2_KT, /*ktHB=*/9,
                  h1seq + (size_t)t * BATCH * H1_STRIDE, H1_STRIDE,
                  hprev, H2_STRIDE,
                  pack2, b2, c2, C2_STRIDE,
                  h2seq + (size_t)t * BATCH * H2_STRIDE, H2_STRIDE,
                  nullptr, 0);
    } else {                                         // ---- layer 3 ----
        int t = s - 2;
        if (t < 0 || t >= TSEQ) return;
        int w3 = w - 400; int nt = w3 / 16, mt = w3 % 16;
        // h3 double buffer: read buf[s&1] (holds h_{t-1}), write buf[(s&1)^1].
        const _Float16* hprev = h3buf + (size_t)(s & 1) * BATCH * H3_STRIDE;
        lstm_tile(mt, nt, U3N, L3_NT, L3_KT, /*ktHB=*/4,
                  h2seq + (size_t)t * BATCH * H2_STRIDE, H2_STRIDE,
                  hprev, H3_STRIDE,
                  pack3, b3, c3, C3_STRIDE,
                  h3buf + (size_t)((s & 1) ^ 1) * BATCH * H3_STRIDE, H3_STRIDE,
                  nullptr, 0);
    }
}

// ---------------------------------------------------------------------------
// Output head: sigmoid(h3_last @ Wd + bd), [256,1] f32.
// ---------------------------------------------------------------------------
__global__ void dense_head(const _Float16* __restrict__ h3,
                           const float* __restrict__ Wd,
                           const float* __restrict__ bd,
                           float* __restrict__ out) {
    int b = blockIdx.x * blockDim.x + threadIdx.x;
    if (b >= BATCH) return;
    float acc = bd[0];
#pragma unroll
    for (int u = 0; u < U3N; ++u)
        acc += (float)h3[(size_t)b * H3_STRIDE + u] * Wd[u];
    out[b] = fsigmoid(acc);
}

// ---------------------------------------------------------------------------
// Host launcher
// ---------------------------------------------------------------------------
extern "C" void kernel_launch(void* const* d_in, const int* in_sizes, int n_in,
                              void* d_out, int out_size, void* d_ws, size_t ws_size,
                              hipStream_t stream) {
    (void)in_sizes; (void)n_in; (void)out_size; (void)ws_size;
    const float* x  = (const float*)d_in[0];
    const float* W1 = (const float*)d_in[1];
    const float* Ur1= (const float*)d_in[2];
    const float* b1 = (const float*)d_in[3];
    const float* W2 = (const float*)d_in[4];
    const float* Ur2= (const float*)d_in[5];
    const float* b2 = (const float*)d_in[6];
    const float* W3 = (const float*)d_in[7];
    const float* Ur3= (const float*)d_in[8];
    const float* b3 = (const float*)d_in[9];
    const float* Wd = (const float*)d_in[10];
    const float* bd = (const float*)d_in[11];
    float* out = (float*)d_out;

    // Workspace carve-up (256B aligned)
    char* ws = (char*)d_ws;
    size_t o = 0;
    auto take = [&](size_t bytes) { size_t r = o; o += (bytes + 255) & ~(size_t)255; return r; };
    size_t off_p1 = take((size_t)PACK1_ELEMS * 2);
    size_t off_p2 = take((size_t)PACK2_ELEMS * 2);
    size_t off_p3 = take((size_t)PACK3_ELEMS * 2);
    size_t zero_begin = o;
    size_t off_h1 = take((size_t)TSEQ * BATCH * H1_STRIDE * 2);
    size_t off_h2 = take((size_t)TSEQ * BATCH * H2_STRIDE * 2);
    size_t off_h3 = take((size_t)2 * BATCH * H3_STRIDE * 2);
    size_t off_zp = take((size_t)BATCH * H1_STRIDE * 2);  // zero pad (never written)
    size_t off_c1 = take((size_t)BATCH * C1_STRIDE * 4);
    size_t off_c2 = take((size_t)BATCH * C2_STRIDE * 4);
    size_t off_c3 = take((size_t)BATCH * C3_STRIDE * 4);
    size_t zero_bytes = o - zero_begin;

    _Float16* p1 = (_Float16*)(ws + off_p1);
    _Float16* p2 = (_Float16*)(ws + off_p2);
    _Float16* p3 = (_Float16*)(ws + off_p3);
    _Float16* h1seq = (_Float16*)(ws + off_h1);
    _Float16* h2seq = (_Float16*)(ws + off_h2);
    _Float16* h3buf = (_Float16*)(ws + off_h3);
    _Float16* zeropad = (_Float16*)(ws + off_zp);
    float* c1 = (float*)(ws + off_c1);
    float* c2 = (float*)(ws + off_c2);
    float* c3 = (float*)(ws + off_c3);

    // 1) Pack weights into WMMA B-fragment order; zero all state/sequence mem.
    pack_l1<<<(PACK1_ELEMS + 255) / 256, 256, 0, stream>>>(W1, Ur1, p1);
    pack_l2<<<(PACK2_ELEMS + 255) / 256, 256, 0, stream>>>(W2, Ur2, p2);
    pack_l3<<<(PACK3_ELEMS + 255) / 256, 256, 0, stream>>>(W3, Ur3, p3);
    zero_ws<<<2048, 256, 0, stream>>>((uint32_t*)(ws + zero_begin), zero_bytes / 4);

    // 2) Pipelined recurrence: 514 super-steps cover t=0..511 for all 3 layers.
    const int blocks = (272 + 128 + 32) * 32 / 256;   // 432 waves -> 54 blocks
    for (int s = 0; s < TSEQ + 2; ++s) {
        lstm_fused_step<<<blocks, 256, 0, stream>>>(
            s, x, b1, b2, b3, p1, p2, p3, h1seq, h2seq, h3buf, zeropad, c1, c2, c3);
    }

    // 3) Head. Last write (s=513, odd) landed in h3buf[0].
    dense_head<<<1, 256, 0, stream>>>(h3buf, Wd, bd, out);
}
